// NSF_Mdiff_CNNcond_3393024164066
// MI455X (gfx1250) — matrix operand order repaired and checked by reading, not compiled, not run on hardware
//
#include <hip/hip_runtime.h>
#include <hip/hip_bf16.h>
#include <math.h>

typedef _Float16 v16h __attribute__((ext_vector_type(16)));
typedef _Float16 v8h  __attribute__((ext_vector_type(8)));
typedef _Float16 v4h  __attribute__((ext_vector_type(4)));
typedef float    v8f  __attribute__((ext_vector_type(8)));

#define NDIM   3
#define KBINS  5
#define TB     3.0f
#define HID    8
#define NCOND  64
#define MIN_BW 0.001f
#define MIN_BH 0.001f
#define MIN_D  0.001f

// Packed GEMM geometry
#define KPAD   96          // padded K (64 cond + x0 + x1 + pad); also hidden-K pad
#define NT12   5           // N-tiles for layers 1/2 (80 cols, 72 used = 9 MLPs * 8)
#define NT3    6           // N-tiles for layer 3 (96 cols, 90 used = 3 dims * 30)
#define NKC    3           // K chunks of 32
#define FRAG_BYTES 1024    // 32 lanes * 32B
#define NFRAG1 (NKC*NT12)  // 15
#define NFRAG2 (NKC*NT12)  // 15
#define NFRAG3 (NKC*NT3)   // 18
#define NFRAGS (NFRAG1+NFRAG2+NFRAG3)  // 48
#define OFF_BIAS1 (NFRAGS*FRAG_BYTES)
#define OFF_BIAS2 (OFF_BIAS1 + 80*4)
#define OFF_BIAS3 (OFF_BIAS2 + 80*4)

#define WAVES   4
#define BLK     128
#define OSTRIDE 99         // LDS row stride (floats) for output transpose, conflict-free

struct Params {
  const float* iW[3][3]; const float* ib[3][3];      // [jd][layer]
  const float* fW[3][2][3]; const float* fb[3][2][3]; // [jd][jf][layer]
};

__device__ __forceinline__ void lds_fence() {
  asm volatile("s_wait_dscnt 0x0" ::: "memory");
}

// ---------------- fast math helpers (native v_exp/v_log/v_rcp paths) --------------------
__device__ __forceinline__ float frcp(float v)  { return __builtin_amdgcn_rcpf(v); }
__device__ __forceinline__ float fexp(float v)  { return __expf(v); }
__device__ __forceinline__ float flog(float v)  { return __logf(v); }
__device__ __forceinline__ float softplusf(float v) {
  // stable, branch-free: max(x,0) + log(1 + exp(-|x|))
  return fmaxf(v, 0.f) + flog(1.f + fexp(-fabsf(v)));
}
__device__ __forceinline__ float tanh_fast(float v) {
  // branch-free: sign(v) * (1 - e^{-2|v|}) / (1 + e^{-2|v|})
  float t = fexp(-2.f * fabsf(v));
  float r = (1.f - t) * frcp(1.f + t);
  return copysignf(r, v);
}

// ---------------- weight-pack entry functions (build block matrices on the fly) ---------
__device__ __forceinline__ float w1e(const Params& P, int k, int col) {
  if (col >= 72) return 0.f;
  int m = col >> 3, u = col & 7;
  int jd = m / 3, t = m % 3;
  if (k >= NCOND + jd || k >= KPAD) return 0.f;
  const float* W = (t == 0) ? P.iW[jd][0] : P.fW[jd][t-1][0];
  return W[k*HID + u];
}
__device__ __forceinline__ float w2e(const Params& P, int k, int col) {
  if (col >= 72 || k >= 72) return 0.f;
  if ((k >> 3) != (col >> 3)) return 0.f;   // block-diagonal per MLP
  int m = col >> 3, jd = m / 3, t = m % 3;
  const float* W = (t == 0) ? P.iW[jd][1] : P.fW[jd][t-1][1];
  return W[(k & 7)*HID + (col & 7)];
}
__device__ __forceinline__ float w3e(const Params& P, int k, int col) {
  if (k >= 72 || col >= 90) return 0.f;
  int mk = k >> 3, u = k & 7;
  int jd = col / 30, c = col % 30;
  int m, oi, od; const float* W;
  if (c < 2)       { m = jd*3;     oi = c;      od = 2;  W = P.iW[jd][2];    }
  else if (c < 16) { m = jd*3 + 1; oi = c - 2;  od = 14; W = P.fW[jd][0][2]; }
  else             { m = jd*3 + 2; oi = c - 16; od = 14; W = P.fW[jd][1][2]; }
  return (mk == m) ? W[u*od + oi] : 0.f;
}
__device__ __forceinline__ float b12e(const Params& P, int col, int layer) {
  if (col >= 72) return 0.f;
  int m = col >> 3, u = col & 7, jd = m / 3, t = m % 3;
  const float* b = (t == 0) ? P.ib[jd][layer] : P.fb[jd][t-1][layer];
  return b[u];
}
__device__ __forceinline__ float b3e(const Params& P, int col) {
  if (col >= 90) return 0.f;
  int jd = col / 30, c = col % 30;
  if (c < 2)  return P.ib[jd][2][c];
  if (c < 16) return P.fb[jd][0][2][c-2];
  return P.fb[jd][1][2][c-16];
}

// ---------------- pack kernel: write B-fragments in WMMA B lane layout, contiguous k order (matches buildA) ------
// B (32x16 f16): lane l holds column n = nt*16 + (l&15); halves h=0..15 hold K = kc*32 + (l>=16?16:0) + h
__global__ __launch_bounds__(256) void nsf_pack(Params P, char* __restrict__ ws) {
  int tid = blockIdx.x*blockDim.x + threadIdx.x;
  if (tid < NFRAGS*32) {
    int f = tid >> 5, lane = tid & 31;
    int k0 = (lane >= 16) ? 16 : 0;
    int kc, nt, layer;
    if (f < NFRAG1)               { layer = 0; kc = f / NT12;        nt = f % NT12; }
    else if (f < NFRAG1+NFRAG2)   { layer = 1; int g = f - NFRAG1;   kc = g / NT12; nt = g % NT12; }
    else                          { layer = 2; int g = f - 30;       kc = g / NT3;  nt = g % NT3;  }
    int n = nt*16 + (lane & 15);
    v16h vals;
    #pragma unroll
    for (int h = 0; h < 16; ++h) {
      int k = kc*32 + k0 + h;
      float v = (layer == 0) ? w1e(P, k, n) : (layer == 1) ? w2e(P, k, n) : w3e(P, k, n);
      vals[h] = (_Float16)v;
    }
    ((v16h*)ws)[f*32 + lane] = vals;
  } else {
    int t = tid - NFRAGS*32;
    if (t < 80)       ((float*)(ws + OFF_BIAS1))[t]       = b12e(P, t, 0);
    else if (t < 160) ((float*)(ws + OFF_BIAS2))[t - 80]  = b12e(P, t - 80, 1);
    else if (t < 256) ((float*)(ws + OFF_BIAS3))[t - 160] = b3e(P, t - 160);
  }
}

// ---------------- main kernel helpers ---------------------------------------------------
// A (16x32 f16) layout: lane l, row m = l&15; lanes<16: halves0-15=K0-15;
// lanes>=16: halves0-15=K16-31.
__device__ __forceinline__ void buildA(const _Float16* fbuf, int mrow, int hi, v16h A[3]) {
  const v8h* rp = (const v8h*)(fbuf + mrow*KPAD);
  #pragma unroll
  for (int kc = 0; kc < NKC; ++kc) {
    v8h lo = rp[kc*4 + 2*hi];
    v8h hh = rp[kc*4 + 2*hi + 1];
    A[kc] = __builtin_shufflevector(lo, hh, 0,1,2,3,4,5,6,7,8,9,10,11,12,13,14,15);
  }
}

// D (16x16 f32) layout: lane l holds column n = l&15; VGPR r holds row m = r + (l>=16 ? 8 : 0)
__device__ __forceinline__ void hidden_layer(const v16h* Bf, int fragBase, const float* biasp,
                                             const v16h A[3], _Float16* fbuf, int lane) {
  int mrow = lane & 15, hi = lane >> 4;
  #pragma unroll
  for (int nt = 0; nt < NT12; ++nt) {
    v8f c = {};
    #pragma unroll
    for (int kc = 0; kc < NKC; ++kc) {
      v16h b = Bf[(fragBase + kc*NT12 + nt)*32 + lane];
      c = __builtin_amdgcn_wmma_f32_16x16x32_f16(false, A[kc], false, b, (short)0, c, false, false);
    }
    float bb = biasp[nt*16 + mrow];
    #pragma unroll
    for (int r = 0; r < 8; ++r) {
      float v = tanh_fast(c[r] + bb);
      fbuf[(hi*8 + r)*KPAD + nt*16 + mrow] = (_Float16)v;
    }
  }
}

__device__ __forceinline__ void out_layer(const v16h* Bf, const float* biasp, const v16h A[3],
                                          float* obuf, int lane, int rowbase) {
  int mrow = lane & 15, hi = lane >> 4;
  #pragma unroll
  for (int nt = 0; nt < NT3; ++nt) {
    v8f c = {};
    #pragma unroll
    for (int kc = 0; kc < NKC; ++kc) {
      v16h b = Bf[(30 + kc*NT3 + nt)*32 + lane];
      c = __builtin_amdgcn_wmma_f32_16x16x32_f16(false, A[kc], false, b, (short)0, c, false, false);
    }
    float bb = biasp[nt*16 + mrow];
    #pragma unroll
    for (int r = 0; r < 8; ++r)
      obuf[(rowbase + hi*8 + r)*OSTRIDE + nt*16 + mrow] = c[r] + bb;
  }
}

__device__ __forceinline__ void stage_input(const float* __restrict__ cond,
                                            const float* __restrict__ x, long long N,
                                            long long srow0, _Float16* fbuf, int lane) {
  int mrow = lane & 15, hi = lane >> 4;
  long long s = srow0 + mrow;
  bool valid = s < N;
  size_t sc = valid ? (size_t)s : 0;
  v4h* rowp = (v4h*)(fbuf + mrow*KPAD);
  const float4* cp = (const float4*)(cond + sc*64);
  if (hi == 0) {
    #pragma unroll
    for (int j = 0; j < 12; ++j) {
      float4 cv = cp[j];
      if (!valid) cv = make_float4(0.f,0.f,0.f,0.f);
      v4h h = { (_Float16)cv.x, (_Float16)cv.y, (_Float16)cv.z, (_Float16)cv.w };
      rowp[j] = h;
    }
  } else {
    #pragma unroll
    for (int j = 12; j < 16; ++j) {
      float4 cv = cp[j];
      if (!valid) cv = make_float4(0.f,0.f,0.f,0.f);
      v4h h = { (_Float16)cv.x, (_Float16)cv.y, (_Float16)cv.z, (_Float16)cv.w };
      rowp[j] = h;
    }
    float x0 = valid ? x[sc*3 + 0] : 0.f;
    float x1 = valid ? x[sc*3 + 1] : 0.f;
    v4h hx = { (_Float16)x0, (_Float16)x1, (_Float16)0.f, (_Float16)0.f };
    rowp[16] = hx;
    v4h zz = {};
    #pragma unroll
    for (int j = 17; j < 24; ++j) rowp[j] = zz;
  }
}

// ---------------- scalar spline math ----------------------------------------------------
__device__ __forceinline__ void softmax5(const float* o, float* r, float scale) {
  float m = o[0];
  #pragma unroll
  for (int j = 1; j < 5; ++j) m = fmaxf(m, o[j]);
  float s = 0.f;
  #pragma unroll
  for (int j = 0; j < 5; ++j) { r[j] = fexp(o[j] - m); s += r[j]; }
  float inv = scale * frcp(s);
  #pragma unroll
  for (int j = 0; j < 5; ++j) r[j] *= inv;
}
template<int NSEL>
__device__ __forceinline__ float sel(const float* a, int idx) {
  float v = a[0];
  #pragma unroll
  for (int j = 1; j < NSEL; ++j) v = (idx == j) ? a[j] : v;
  return v;
}
// reference-exact structure: second softmax inside _rqs_fwd, affine to [-TB,TB], endpoints pinned
__device__ __forceinline__ void build_knots(const float* u5, float minb, float* cum, float* len) {
  float sm[5];
  softmax5(u5, sm, 1.f);
  cum[0] = -TB;
  float run = 0.f;
  #pragma unroll
  for (int j = 0; j < 5; ++j) {
    float w = minb + (1.f - 5.f*minb)*sm[j];
    run += w;
    cum[j+1] = 2.f*TB*run - TB;
  }
  cum[5] = TB;
  #pragma unroll
  for (int j = 0; j < 5; ++j) len[j] = cum[j+1] - cum[j];
}

// ---------------- main kernel -----------------------------------------------------------
__global__ __launch_bounds__(BLK) void nsf_main(
    const float* __restrict__ x, const float* __restrict__ cond,
    const float* __restrict__ mask, const char* __restrict__ ws,
    float* __restrict__ out, int N, int NP)
{
  __shared__ alignas(16) _Float16 s_f[WAVES][16*KPAD];
  __shared__ float s_o[WAVES][32*OSTRIDE];
  const int lane = threadIdx.x & 31;
  const int w    = threadIdx.x >> 5;
  const int gw   = blockIdx.x*WAVES + w;
  const int nw   = gridDim.x*WAVES;
  const v16h* Bf = (const v16h*)ws;
  const float* b1p = (const float*)(ws + OFF_BIAS1);
  const float* b2p = (const float*)(ws + OFF_BIAS2);
  const float* b3p = (const float*)(ws + OFF_BIAS3);
  _Float16* fbuf = s_f[w];
  float*    obuf = s_o[w];
  const float dconst = 0.5397422f;  // log(exp(1 - MIN_D) - 1)

  for (int p = gw; p < NP; p += nw) {
    long long base = (long long)p * 32;
    if (p + nw < NP)
      __builtin_prefetch(cond + (size_t)(p + nw)*32*64 + (size_t)lane*64, 0, 1);

    #pragma unroll
    for (int th = 0; th < 2; ++th) {
      stage_input(cond, x, (long long)N, base + th*16, fbuf, lane);
      lds_fence();
      v16h A[3];
      buildA(fbuf, lane & 15, lane >> 4, A);
      hidden_layer(Bf, 0, b1p, A, fbuf, lane);        // L1: cond -> h1 (tanh)
      lds_fence();
      buildA(fbuf, lane & 15, lane >> 4, A);
      hidden_layer(Bf, NFRAG1, b2p, A, fbuf, lane);   // L2: h1 -> h2 (tanh)
      lds_fence();
      buildA(fbuf, lane & 15, lane >> 4, A);
      out_layer(Bf, b3p, A, obuf, lane, th*16);       // L3: h2 -> 90 outputs
      lds_fence();
    }

    // ---------- elementwise spline + logp: lane = sample ----------
    long long s = base + lane;
    if (s < (long long)N) {
      const float* orow = obuf + lane*OSTRIDE;
      float logp = 0.f;
      #pragma unroll
      for (int jd = 0; jd < NDIM; ++jd) {
        const float* od = orow + jd*30;
        float mu = od[0], alpha = od[1];
        float inv_sig = fexp(-alpha);          // 1/sigma; log(sigma) == alpha exactly
        float xv = x[(size_t)s*3 + jd];
        float ld_tot = 0.f;
        #pragma unroll
        for (int jf = 0; jf < 2; ++jf) {
          const float* oo = od + 2 + jf*14;
          float uw[5], uh[5];
          softmax5(oo,     uw, 2.f*TB);   // W = 2B*softmax(o[:K])
          softmax5(oo + 5, uh, 2.f*TB);   // H = 2B*softmax(o[K:2K])
          float ud6[6];
          ud6[0] = dconst; ud6[5] = dconst;
          #pragma unroll
          for (int i = 0; i < 4; ++i) ud6[i+1] = softplusf(oo[10 + i]);  // D = softplus(o[2K:])
          bool inside = (xv >= -TB) && (xv <= TB);
          float xc = fminf(fmaxf(xv, -TB), TB);
          float cw[6], wid[5], ch[6], hei[5];
          build_knots(uw, MIN_BW, cw, wid);
          build_knots(uh, MIN_BH, ch, hei);
          float dv[6];
          #pragma unroll
          for (int i = 0; i < 6; ++i) dv[i] = MIN_D + softplusf(ud6[i]);
          int cnt = 0;
          #pragma unroll
          for (int j = 0; j < 6; ++j) {
            float e = (j == 5) ? (TB + 1e-6f) : cw[j];
            cnt += (xc >= e) ? 1 : 0;
          }
          int idx = cnt - 1; idx = idx < 0 ? 0 : (idx > 4 ? 4 : idx);
          float in_cumw  = sel<5>(cw,  idx);
          float in_w     = sel<5>(wid, idx);
          float in_cumh  = sel<5>(ch,  idx);
          float in_h     = sel<5>(hei, idx);
          float rw       = frcp(in_w);
          float in_delta = in_h * rw;
          float d_k   = sel<5>(dv,     idx);
          float d_kp1 = sel<5>(dv + 1, idx);
          float theta = (xc - in_cumw) * rw;
          float t1mt  = theta * (1.f - theta);
          float numer = in_h * (in_delta*theta*theta + d_k*t1mt);
          float denom = in_delta + (d_k + d_kp1 - 2.f*in_delta)*t1mt;
          float rd    = frcp(denom);
          float outv  = in_cumh + numer*rd;
          float om    = 1.f - theta;
          float dnum  = in_delta*in_delta*(d_kp1*theta*theta + 2.f*in_delta*t1mt + d_k*om*om);
          float ld    = flog(dnum*rd*rd);   // log(dnum) - 2*log(denom)
          xv     = inside ? outv : xv;
          ld_tot += inside ? ld : 0.f;
        }
        float zz = (xv - mu) * inv_sig;
        float lp = -(alpha + zz + fexp(-zz));
        if (__builtin_isnan(lp) || __builtin_isinf(lp)) lp = -100.f;
        logp += (ld_tot + lp) * mask[(size_t)s*3 + jd];
      }
      out[s] = logp;
    }
    lds_fence();
  }
}

// ---------------- host launcher ---------------------------------------------------------
extern "C" void kernel_launch(void* const* d_in, const int* in_sizes, int n_in,
                              void* d_out, int out_size, void* d_ws, size_t ws_size,
                              hipStream_t stream) {
  const float* x    = (const float*)d_in[0];
  const float* cond = (const float*)d_in[1];
  const float* mask = (const float*)d_in[2];
  int N = in_sizes[0] / 3;

  Params P;
  for (int jd = 0; jd < 3; ++jd) {
    int b = 3 + jd*6;
    for (int l = 0; l < 3; ++l) {
      P.iW[jd][l] = (const float*)d_in[b + l*2 + 0];
      P.ib[jd][l] = (const float*)d_in[b + l*2 + 1];
    }
    for (int jf = 0; jf < 2; ++jf) {
      int c = 21 + (jd*2 + jf)*6;
      for (int l = 0; l < 3; ++l) {
        P.fW[jd][jf][l] = (const float*)d_in[c + l*2 + 0];
        P.fb[jd][jf][l] = (const float*)d_in[c + l*2 + 1];
      }
    }
  }

  char* ws = (char*)d_ws;
  hipLaunchKernelGGL(nsf_pack, dim3(7), dim3(256), 0, stream, P, ws);

  int NP = (N + 31) / 32;
  int blocks = 2048;
  if (blocks*WAVES > NP) blocks = (NP + WAVES - 1) / WAVES;
  if (blocks < 1) blocks = 1;
  hipLaunchKernelGGL(nsf_main, dim3(blocks), dim3(BLK), 0, stream,
                     x, cond, mask, (const char*)ws, (float*)d_out, N, NP);
}
